// Encoder_Rel_64046552318121
// MI455X (gfx1250) — compile-verified, matrix operand order checked by reading, not run on hardware
//
#include <hip/hip_runtime.h>
#include <hip/hip_bf16.h>

#define R_N 40000
#define E_N 500000

typedef __attribute__((ext_vector_type(16))) __bf16 v16bf;
typedef __attribute__((ext_vector_type(8)))  float  v8f;

__device__ __forceinline__ float lk(float x) { return x > 0.f ? x : 0.2f * x; }

// monotonic order-preserving float<->uint encoding for atomic max
__device__ __forceinline__ unsigned fenc(float x) {
    unsigned u = __float_as_uint(x);
    return (u & 0x80000000u) ? ~u : (u | 0x80000000u);
}
__device__ __forceinline__ float fdec(unsigned k) {
    return (k & 0x80000000u) ? __uint_as_float(k & 0x7fffffffu) : __uint_as_float(~k);
}

__device__ __forceinline__ void atomAddF(float* p, float v) {
    __hip_atomic_fetch_add(p, v, __ATOMIC_RELAXED, __HIP_MEMORY_SCOPE_AGENT);
}
__device__ __forceinline__ void atomMaxU(unsigned* p, unsigned v) {
    __hip_atomic_fetch_max(p, v, __ATOMIC_RELAXED, __HIP_MEMORY_SCOPE_AGENT);
}

__device__ __forceinline__ void pack16(v16bf& d, float4 f0, float4 f1, float4 f2, float4 f3) {
    d[0]  = (__bf16)f0.x; d[1]  = (__bf16)f0.y; d[2]  = (__bf16)f0.z; d[3]  = (__bf16)f0.w;
    d[4]  = (__bf16)f1.x; d[5]  = (__bf16)f1.y; d[6]  = (__bf16)f1.z; d[7]  = (__bf16)f1.w;
    d[8]  = (__bf16)f2.x; d[9]  = (__bf16)f2.y; d[10] = (__bf16)f2.z; d[11] = (__bf16)f2.w;
    d[12] = (__bf16)f3.x; d[13] = (__bf16)f3.y; d[14] = (__bf16)f3.z; d[15] = (__bf16)f3.w;
}

// -------------------- WMMA GEMM: Out[R,64] = act(A[R,64] @ W[64,64] + bias) (+ relu(addsrc)) -----
// block = 128 threads (4 waves); wave w computes the 16x16 tile at cols [16w,16w+16)
template <int ACT, bool HAS_BIAS, bool HAS_ADD>
__global__ __launch_bounds__(128)
void gemm64_wmma(const float* __restrict__ A, const float* __restrict__ W,
                 const float* __restrict__ bias, const float* __restrict__ addsrc,
                 float* __restrict__ Out) {
    __shared__ float ldsW[64 * 64];

    // cooperative coalesced stage of W into LDS (128 threads x 32 floats = 4096)
    {
        const float4* Wg = (const float4*)W;
        float4*       Wl = (float4*)ldsW;
#pragma unroll
        for (int i = 0; i < 8; ++i)
            Wl[threadIdx.x + i * 128] = Wg[threadIdx.x + i * 128];
    }
    __syncthreads();

    const int lane    = threadIdx.x & 31;
    const int wave    = threadIdx.x >> 5;
    const int rowBase = blockIdx.x * 16;
    const int l16     = lane & 15;
    const int hi      = lane >> 4;          // 0 or 1
    const int khalf   = hi * 8;             // K sub-offset per 16-bit A/B layout
    const int row     = rowBase + l16;
    const int col     = wave * 16 + l16;

    // A: 4x contiguous 8-float chunks per lane -> 8x b128 loads
    const float4* Ar4 = (const float4*)(A + (size_t)row * 64);
    const int c0 = khalf >> 2;              // 0 or 2
    float4 f0 = Ar4[c0 + 0],  f1 = Ar4[c0 + 1];
    float4 f2 = Ar4[c0 + 4],  f3 = Ar4[c0 + 5];
    float4 f4 = Ar4[c0 + 8],  f5 = Ar4[c0 + 9];
    float4 f6 = Ar4[c0 + 12], f7 = Ar4[c0 + 13];

    v16bf a0, a1, b0, b1;
    pack16(a0, f0, f1, f2, f3);
    pack16(a1, f4, f5, f6, f7);

#pragma unroll
    for (int i = 0; i < 8; ++i) {
        b0[i]     = (__bf16)ldsW[(khalf + i) * 64 + col];
        b0[i + 8] = (__bf16)ldsW[(16 + khalf + i) * 64 + col];
        b1[i]     = (__bf16)ldsW[(32 + khalf + i) * 64 + col];
        b1[i + 8] = (__bf16)ldsW[(48 + khalf + i) * 64 + col];
    }

    v8f c = {0.f, 0.f, 0.f, 0.f, 0.f, 0.f, 0.f, 0.f};
    c = __builtin_amdgcn_wmma_f32_16x16x32_bf16(false, a0, false, b0, (short)0, c, false, false);
    c = __builtin_amdgcn_wmma_f32_16x16x32_bf16(false, a1, false, b1, (short)0, c, false, false);

    const float bcol = HAS_BIAS ? bias[col] : 0.f;
#pragma unroll
    for (int j = 0; j < 8; ++j) {
        const int m = rowBase + j + hi * 8;          // C/D layout: VGPR j -> M = j (+8 for hi lanes)
        float v = c[j] + bcol;
        if (ACT == 1) v = fmaxf(v, 0.f);
        if (HAS_ADD)  v += fmaxf(addsrc[(size_t)m * 64 + col], 0.f);
        Out[(size_t)m * 64 + col] = v;
    }
}

// -------------------- zero scratch --------------------
__global__ void zero_kernel(float* __restrict__ p, int n) {
    int i = blockIdx.x * blockDim.x + threadIdx.x;
    if (i < n) p[i] = 0.f;
}

// -------------------- pass 1: raw attention logits + segment max --------------------
// one thread per (edge, head-group); 8 consecutive lanes share an edge -> coalesced row gathers
__global__ __launch_bounds__(256)
void edge_raw_kernel(const int* __restrict__ trip, const float* __restrict__ Ph,
                     const float* __restrict__ Pt, const float* __restrict__ avec,
                     const float* __restrict__ abin, float* __restrict__ rawbuf,
                     unsigned* __restrict__ menc) {
    int t = blockIdx.x * blockDim.x + threadIdx.x;
    int e = t >> 3, h = t & 7;
    if (e >= E_N) return;
    int head = trip[3 * e], tail = trip[3 * e + 1], bin = trip[3 * e + 2];

    const float4* ph = (const float4*)(Ph + (size_t)head * 64 + h * 8);
    const float4* pt = (const float4*)(Pt + (size_t)tail * 64 + h * 8);
    const float4* av = (const float4*)(avec + h * 8);
    float4 p0 = ph[0], p1 = ph[1];
    float4 q0 = pt[0], q1 = pt[1];
    float4 v0 = av[0], v1 = av[1];

    float acc = lk(p0.x + q0.x) * v0.x + lk(p0.y + q0.y) * v0.y +
                lk(p0.z + q0.z) * v0.z + lk(p0.w + q0.w) * v0.w +
                lk(p1.x + q1.x) * v1.x + lk(p1.y + q1.y) * v1.y +
                lk(p1.z + q1.z) * v1.z + lk(p1.w + q1.w) * v1.w;

    float raw = acc + lk(abin[bin * 8 + h]);
    rawbuf[(size_t)e * 8 + h] = raw;
    atomMaxU(&menc[head * 8 + h], fenc(raw));
}

// -------------------- pass 2: exp(raw - max), segment sum --------------------
__global__ __launch_bounds__(256)
void edge_exp_kernel(const int* __restrict__ trip, float* __restrict__ ebuf,
                     const unsigned* __restrict__ menc, float* __restrict__ sbuf) {
    int t = blockIdx.x * blockDim.x + threadIdx.x;
    int e = t >> 3, h = t & 7;
    if (e >= E_N) return;
    int head = trip[3 * e];
    float m  = fdec(menc[head * 8 + h]);
    float ev = __expf(ebuf[(size_t)e * 8 + h] - m);
    ebuf[(size_t)e * 8 + h] = ev;
    atomAddF(&sbuf[head * 8 + h], ev);
}

// -------------------- pass 3: beta-weighted message scatter --------------------
__global__ __launch_bounds__(256)
void edge_scatter_kernel(const int* __restrict__ trip, const float* __restrict__ ebuf,
                         const float* __restrict__ sbuf, const float* __restrict__ agg,
                         float* __restrict__ accum) {
    int t = blockIdx.x * blockDim.x + threadIdx.x;
    int e = t >> 3, h = t & 7;
    if (e >= E_N) return;
    int head = trip[3 * e], tail = trip[3 * e + 1];
    float beta = ebuf[(size_t)e * 8 + h] / (sbuf[head * 8 + h] + 1e-16f);

    const float4* ag = (const float4*)(agg + (size_t)tail * 64 + h * 8);
    float4 g0 = ag[0], g1 = ag[1];
    float* dst = accum + (size_t)head * 64 + h * 8;
    atomAddF(dst + 0, g0.x * beta);
    atomAddF(dst + 1, g0.y * beta);
    atomAddF(dst + 2, g0.z * beta);
    atomAddF(dst + 3, g0.w * beta);
    atomAddF(dst + 4, g1.x * beta);
    atomAddF(dst + 5, g1.y * beta);
    atomAddF(dst + 6, g1.z * beta);
    atomAddF(dst + 7, g1.w * beta);
}

extern "C" void kernel_launch(void* const* d_in, const int* in_sizes, int n_in,
                              void* d_out, int out_size, void* d_ws, size_t ws_size,
                              hipStream_t stream) {
    const int*   trip        = (const int*)d_in[0];     // [E,3] int32
    const float* rel_emb_w   = (const float*)d_in[1];   // [R,64]
    const float* proj1_w     = (const float*)d_in[2];   // [64,64]
    const float* proj1_b     = (const float*)d_in[3];   // [64]
    const float* attn_proj_w = (const float*)d_in[4];   // [2,128,64]
    const float* attn_proj_b = (const float*)d_in[5];   // [2,64]
    const float* attn_bin    = (const float*)d_in[6];   // [2,10,8,1]
    const float* attn_vec    = (const float*)d_in[7];   // [2,8,8]
    const float* aggr_w      = (const float*)d_in[8];   // [2,64,64]
    const float* aggr_b      = (const float*)d_in[9];   // [2,64]
    const float* res_w       = (const float*)d_in[10];  // [2,64,64]
    const float* res_b       = (const float*)d_in[11];  // [2,64]
    float* out = (float*)d_out;

    const size_t R64 = (size_t)R_N * 64, R8 = (size_t)R_N * 8;
    float* ws    = (float*)d_ws;
    float* emb0  = ws;                 // [R,64]
    float* emb1  = emb0 + R64;         // [R,64]
    float* Ph    = emb1 + R64;         // [R,64]
    float* Pt    = Ph + R64;           // [R,64]
    float* agg   = Pt + R64;           // [R,64]
    float* accum = agg + R64;          // [R,64]   (accum, sbuf, menc contiguous -> one zero pass)
    float* sbuf  = accum + R64;        // [R,8]
    unsigned* menc = (unsigned*)(sbuf + R8); // [R,8]
    float* ebuf  = (float*)(menc + R8);      // [E,8]

    const dim3 gB(128), gG(R_N / 16);
    const int e8       = E_N * 8;
    const int eBlocks  = (e8 + 255) / 256;
    const int zN       = (int)(R64 + 2 * R8);   // accum + sbuf + menc
    const int zBlocks  = (zN + 255) / 256;

    // emb0 = relu(rel_emb_w @ proj1_w + proj1_b)
    gemm64_wmma<1, true, false><<<gG, gB, 0, stream>>>(rel_emb_w, proj1_w, proj1_b, nullptr, emb0);

    float* emb = emb0;
    for (int l = 0; l < 2; ++l) {
        const float* pw   = attn_proj_w + (size_t)l * 128 * 64;
        const float* pb   = attn_proj_b + (size_t)l * 64;
        const float* abin = attn_bin   + (size_t)l * 10 * 8;
        const float* avec = attn_vec   + (size_t)l * 64;
        const float* aw   = aggr_w     + (size_t)l * 64 * 64;
        const float* ab   = aggr_b     + (size_t)l * 64;
        const float* rw   = res_w      + (size_t)l * 64 * 64;
        const float* rb   = res_b      + (size_t)l * 64;

        // Hoisted per-relation GEMMs (replaces per-edge GEMMs of the reference)
        gemm64_wmma<0, false, false><<<gG, gB, 0, stream>>>(emb, pw,           nullptr, nullptr, Ph);
        gemm64_wmma<0, true,  false><<<gG, gB, 0, stream>>>(emb, pw + 64 * 64, pb,      nullptr, Pt);
        gemm64_wmma<0, true,  false><<<gG, gB, 0, stream>>>(emb, aw,           ab,      nullptr, agg);

        zero_kernel<<<zBlocks, 256, 0, stream>>>(accum, zN);

        edge_raw_kernel    <<<eBlocks, 256, 0, stream>>>(trip, Ph, Pt, avec, abin, ebuf, menc);
        edge_exp_kernel    <<<eBlocks, 256, 0, stream>>>(trip, ebuf, menc, sbuf);
        edge_scatter_kernel<<<eBlocks, 256, 0, stream>>>(trip, ebuf, sbuf, agg, accum);

        // emb_next = relu(accum) + relu(emb @ rw + rb)
        float* dst = (l == 1) ? out : emb1;
        gemm64_wmma<1, true, true><<<gG, gB, 0, stream>>>(emb, rw, rb, accum, dst);
        emb = dst;
    }
}
